// DynamicTriModalFusion_25323127177294
// MI455X (gfx1250) — matrix operand order repaired and checked by reading, not compiled, not run on hardware
//
#include <hip/hip_runtime.h>

#define BT   32768     // B*T
#define DDIM 512

typedef __attribute__((ext_vector_type(16))) __bf16       v16bf;
typedef __attribute__((ext_vector_type(8)))  float        v8f;
typedef __attribute__((ext_vector_type(8)))  unsigned int u32x8;

__device__ __forceinline__ unsigned short bf16_rne(float f) {
    unsigned int u = __builtin_bit_cast(unsigned int, f);
    return (unsigned short)((u + 0x7FFFu + ((u >> 16) & 1u)) >> 16);
}
__device__ __forceinline__ float bf16_to_f32(unsigned short h) {
    unsigned int u = ((unsigned int)h) << 16;
    return __builtin_bit_cast(float, u);
}

// Load one v16bf WMMA fragment = two 16B LDS reads (delta in ushort units).
__device__ __forceinline__ v16bf load_frag(const unsigned short* p, int delta) {
    uint4 a = *(const uint4*)p;
    uint4 b = *(const uint4*)(p + delta);
    u32x8 r;
    r[0] = a.x; r[1] = a.y; r[2] = a.z; r[3] = a.w;
    r[4] = b.x; r[5] = b.y; r[6] = b.z; r[7] = b.w;
    return __builtin_bit_cast(v16bf, r);
}

__device__ __forceinline__ void softmax3_store(float a, float b, float c, float* dst) {
    float m  = fmaxf(a, fmaxf(b, c));
    float ea = __expf(a - m), eb = __expf(b - m), ec = __expf(c - m);
    float inv = 1.0f / (ea + eb + ec);
    dst[0] = ea * inv; dst[1] = eb * inv; dst[2] = ec * inv;
}

// ---------------------------------------------------------------------------
// Kernel 1: pairwise cosine sims + softmax weights. One wave32 per (b,t) row.
// wts layout: [3][BT][3]  (== final weights output layout [3,B,T,3])
// ---------------------------------------------------------------------------
__global__ __launch_bounds__(256) void trimodal_weights_kernel(
    const float* __restrict__ s0, const float* __restrict__ s1,
    const float* __restrict__ s2, float* __restrict__ wts)
{
    const int lane = threadIdx.x & 31;
    const int wave = threadIdx.x >> 5;
    const size_t row = (size_t)blockIdx.x * 8 + wave;
    const float* p0 = s0 + row * DDIM;
    const float* p1 = s1 + row * DDIM;
    const float* p2 = s2 + row * DDIM;

    float d00 = 0.f, d01 = 0.f, d02 = 0.f, d11 = 0.f, d12 = 0.f, d22 = 0.f;
    #pragma unroll
    for (int c = 0; c < 4; ++c) {
        int off = c * 128 + lane * 4;
        float4 a = *(const float4*)(p0 + off);
        float4 b = *(const float4*)(p1 + off);
        float4 g = *(const float4*)(p2 + off);
        d00 += a.x*a.x + a.y*a.y + a.z*a.z + a.w*a.w;
        d11 += b.x*b.x + b.y*b.y + b.z*b.z + b.w*b.w;
        d22 += g.x*g.x + g.y*g.y + g.z*g.z + g.w*g.w;
        d01 += a.x*b.x + a.y*b.y + a.z*b.z + a.w*b.w;
        d02 += a.x*g.x + a.y*g.y + a.z*g.z + a.w*g.w;
        d12 += b.x*g.x + b.y*g.y + b.z*g.z + b.w*g.w;
    }
    #pragma unroll
    for (int o = 16; o > 0; o >>= 1) {
        d00 += __shfl_xor(d00, o); d01 += __shfl_xor(d01, o);
        d02 += __shfl_xor(d02, o); d11 += __shfl_xor(d11, o);
        d12 += __shfl_xor(d12, o); d22 += __shfl_xor(d22, o);
    }
    if (lane == 0) {
        float n0 = sqrtf(d00), n1 = sqrtf(d11), n2 = sqrtf(d22);
        const float it = 2.0f;  // 1/TEMPERATURE
        float s00 = d00 / fmaxf(n0*n0, 1e-8f) * it;
        float s01 = d01 / fmaxf(n0*n1, 1e-8f) * it;
        float s02 = d02 / fmaxf(n0*n2, 1e-8f) * it;
        float s11 = d11 / fmaxf(n1*n1, 1e-8f) * it;
        float s12 = d12 / fmaxf(n1*n2, 1e-8f) * it;
        float s22 = d22 / fmaxf(n2*n2, 1e-8f) * it;
        float w[3];
        softmax3_store(s00, s01, s02, w);
        wts[((size_t)0*BT + row)*3 + 0] = w[0];
        wts[((size_t)0*BT + row)*3 + 1] = w[1];
        wts[((size_t)0*BT + row)*3 + 2] = w[2];
        softmax3_store(s01, s11, s12, w);
        wts[((size_t)1*BT + row)*3 + 0] = w[0];
        wts[((size_t)1*BT + row)*3 + 1] = w[1];
        wts[((size_t)1*BT + row)*3 + 2] = w[2];
        softmax3_store(s02, s12, s22, w);
        wts[((size_t)2*BT + row)*3 + 0] = w[0];
        wts[((size_t)2*BT + row)*3 + 1] = w[1];
        wts[((size_t)2*BT + row)*3 + 2] = w[2];
    }
}

// ---------------------------------------------------------------------------
// Kernel 2: fused = sum_j w_j*S_j built on the fly; p = fused @ W^T + b;
// x = S_i + p written to xout.  128x128 tile, 8 wave32, bf16 hi/lo split WMMA.
// ---------------------------------------------------------------------------
__global__ __launch_bounds__(256) void trimodal_gemm_kernel(
    const float* __restrict__ s0, const float* __restrict__ s1,
    const float* __restrict__ s2, const float* __restrict__ pw,
    const float* __restrict__ pb, const float* __restrict__ wts,
    float* __restrict__ xout)
{
    constexpr int LDT = 40;  // padded pitch in bf16: 80B = 20 banks, 16B aligned
    __shared__ unsigned short Ah[128 * LDT];
    __shared__ unsigned short Al[128 * LDT];
    __shared__ unsigned short Bh[128 * LDT];
    __shared__ unsigned short Bl[128 * LDT];
    __shared__ float wl[128 * 3];

    const int i  = blockIdx.z;
    const int m0 = blockIdx.x * 128;
    const int n0 = blockIdx.y * 128;
    const int t  = threadIdx.x;
    const int lane  = t & 31;
    const int wid   = t >> 5;
    const int waveM = wid & 3;    // 4 row groups of 32
    const int waveN = wid >> 2;   // 2 col groups of 64
    const int fr = lane & 15;     // frag M-row (A) / N-col (B,D)
    const int hb = lane >> 4;     // half-wave

    const float* Si = (i == 0) ? s0 : (i == 1 ? s1 : s2);
    const float* W  = pw + (size_t)i * DDIM * DDIM;

    if (t < 128) {
        size_t wb = ((size_t)i * BT + m0 + t) * 3;
        wl[t*3+0] = wts[wb+0];
        wl[t*3+1] = wts[wb+1];
        wl[t*3+2] = wts[wb+2];
    }

    v8f acc[2][4];
    #pragma unroll
    for (int mf = 0; mf < 2; ++mf)
        #pragma unroll
        for (int nf = 0; nf < 4; ++nf)
            #pragma unroll
            for (int e = 0; e < 8; ++e) acc[mf][nf][e] = 0.0f;

    for (int kt = 0; kt < 16; ++kt) {
        const int k0 = kt * 32;
        __syncthreads();
        // ---- A tile: fused rows, bf16 hi/lo ----
        #pragma unroll
        for (int p = 0; p < 4; ++p) {
            int idx = t + 256 * p;
            int r   = idx >> 3;
            int c4  = (idx & 7) * 4;
            size_t go = (size_t)(m0 + r) * DDIM + k0 + c4;
            float4 a = *(const float4*)(s0 + go);
            float4 b = *(const float4*)(s1 + go);
            float4 g = *(const float4*)(s2 + go);
            float w0 = wl[r*3+0], w1 = wl[r*3+1], w2 = wl[r*3+2];
            float f0 = w0*a.x + w1*b.x + w2*g.x;
            float f1 = w0*a.y + w1*b.y + w2*g.y;
            float f2 = w0*a.z + w1*b.z + w2*g.z;
            float f3 = w0*a.w + w1*b.w + w2*g.w;
            unsigned short h0 = bf16_rne(f0), h1 = bf16_rne(f1);
            unsigned short h2 = bf16_rne(f2), h3 = bf16_rne(f3);
            unsigned short l0 = bf16_rne(f0 - bf16_to_f32(h0));
            unsigned short l1 = bf16_rne(f1 - bf16_to_f32(h1));
            unsigned short l2 = bf16_rne(f2 - bf16_to_f32(h2));
            unsigned short l3 = bf16_rne(f3 - bf16_to_f32(h3));
            int lo_ = r * LDT + c4;
            uint2 uh; uh.x = (unsigned)h0 | ((unsigned)h1 << 16);
                      uh.y = (unsigned)h2 | ((unsigned)h3 << 16);
            uint2 ul; ul.x = (unsigned)l0 | ((unsigned)l1 << 16);
                      ul.y = (unsigned)l2 | ((unsigned)l3 << 16);
            *(uint2*)&Ah[lo_] = uh;
            *(uint2*)&Al[lo_] = ul;
        }
        // ---- B tile: W[n][k] (== W^T[k][n]), bf16 hi/lo ----
        #pragma unroll
        for (int p = 0; p < 4; ++p) {
            int idx = t + 256 * p;
            int n   = idx >> 3;
            int c4  = (idx & 7) * 4;
            float4 wv = *(const float4*)(W + (size_t)(n0 + n) * DDIM + k0 + c4);
            unsigned short h0 = bf16_rne(wv.x), h1 = bf16_rne(wv.y);
            unsigned short h2 = bf16_rne(wv.z), h3 = bf16_rne(wv.w);
            unsigned short l0 = bf16_rne(wv.x - bf16_to_f32(h0));
            unsigned short l1 = bf16_rne(wv.y - bf16_to_f32(h1));
            unsigned short l2 = bf16_rne(wv.z - bf16_to_f32(h2));
            unsigned short l3 = bf16_rne(wv.w - bf16_to_f32(h3));
            int lo_ = n * LDT + c4;
            uint2 uh; uh.x = (unsigned)h0 | ((unsigned)h1 << 16);
                      uh.y = (unsigned)h2 | ((unsigned)h3 << 16);
            uint2 ul; ul.x = (unsigned)l0 | ((unsigned)l1 << 16);
                      ul.y = (unsigned)l2 | ((unsigned)l3 << 16);
            *(uint2*)&Bh[lo_] = uh;
            *(uint2*)&Bl[lo_] = ul;
        }
        if (kt < 15 && t < 128) {   // prefetch next K chunk (global_prefetch_b8)
            int nk = k0 + 32;
            __builtin_prefetch(W  + (size_t)(n0 + t) * DDIM + nk, 0, 3);
            __builtin_prefetch(s0 + (size_t)(m0 + t) * DDIM + nk, 0, 3);
            __builtin_prefetch(s1 + (size_t)(m0 + t) * DDIM + nk, 0, 3);
            __builtin_prefetch(s2 + (size_t)(m0 + t) * DDIM + nk, 0, 3);
        }
        __syncthreads();
        // ---- WMMA: acc += Ahi*Bhi + Ahi*Blo + Alo*Bhi ----
        v16bf a_hi[2], a_lo[2];
        #pragma unroll
        for (int mf = 0; mf < 2; ++mf) {
            int base = (waveM*32 + mf*16 + fr) * LDT + hb * 8;
            a_hi[mf] = load_frag(&Ah[base], 16);
            a_lo[mf] = load_frag(&Al[base], 16);
        }
        #pragma unroll
        for (int nf = 0; nf < 4; ++nf) {
            int base = (waveN*64 + nf*16 + fr) * LDT + hb * 8;
            v16bf b_hi = load_frag(&Bh[base], 16);
            v16bf b_lo = load_frag(&Bl[base], 16);
            #pragma unroll
            for (int mf = 0; mf < 2; ++mf) {
                acc[mf][nf] = __builtin_amdgcn_wmma_f32_16x16x32_bf16(
                    false, a_hi[mf], false, b_hi, (short)0, acc[mf][nf], false, false);
                acc[mf][nf] = __builtin_amdgcn_wmma_f32_16x16x32_bf16(
                    false, a_hi[mf], false, b_lo, (short)0, acc[mf][nf], false, false);
                acc[mf][nf] = __builtin_amdgcn_wmma_f32_16x16x32_bf16(
                    false, a_lo[mf], false, b_hi, (short)0, acc[mf][nf], false, false);
            }
        }
    }

    // ---- epilogue: x = S_i + (p + bias) ----
    #pragma unroll
    for (int nf = 0; nf < 4; ++nf) {
        int n = n0 + waveN*64 + nf*16 + fr;
        float bias = pb[i * DDIM + n];
        #pragma unroll
        for (int mf = 0; mf < 2; ++mf) {
            #pragma unroll
            for (int r = 0; r < 8; ++r) {
                int m = m0 + waveM*32 + mf*16 + hb*8 + r;
                xout[((size_t)i * BT + m) * DDIM + n] =
                    acc[mf][nf][r] + bias + Si[(size_t)m * DDIM + n];
            }
        }
    }
}

// ---------------------------------------------------------------------------
// Kernel 3: in-place LayerNorm over D per row. One wave32 per row.
// ---------------------------------------------------------------------------
__global__ __launch_bounds__(256) void trimodal_ln_kernel(
    float* __restrict__ x, const float* __restrict__ g, const float* __restrict__ b)
{
    const int lane = threadIdx.x & 31;
    const int wave = threadIdx.x >> 5;
    const size_t row = (size_t)blockIdx.x * 8 + wave;   // 0..3*BT-1
    const int i = (int)(row / BT);
    float* p = x + row * DDIM;

    float4 v[4];
    float sum = 0.f, sq = 0.f;
    #pragma unroll
    for (int c = 0; c < 4; ++c) {
        v[c] = *(const float4*)(p + c*128 + lane*4);
        sum += v[c].x + v[c].y + v[c].z + v[c].w;
        sq  += v[c].x*v[c].x + v[c].y*v[c].y + v[c].z*v[c].z + v[c].w*v[c].w;
    }
    #pragma unroll
    for (int o = 16; o > 0; o >>= 1) {
        sum += __shfl_xor(sum, o);
        sq  += __shfl_xor(sq, o);
    }
    float mu  = sum * (1.0f / DDIM);
    float var = sq * (1.0f / DDIM) - mu * mu;
    float inv = rsqrtf(var + 1e-5f);
    #pragma unroll
    for (int c = 0; c < 4; ++c) {
        int off = c*128 + lane*4;
        float4 gg = *(const float4*)(g + i*DDIM + off);
        float4 bb = *(const float4*)(b + i*DDIM + off);
        float4 o;
        o.x = (v[c].x - mu) * inv * gg.x + bb.x;
        o.y = (v[c].y - mu) * inv * gg.y + bb.y;
        o.z = (v[c].z - mu) * inv * gg.z + bb.z;
        o.w = (v[c].w - mu) * inv * gg.w + bb.w;
        *(float4*)(p + off) = o;
    }
}

extern "C" void kernel_launch(void* const* d_in, const int* in_sizes, int n_in,
                              void* d_out, int out_size, void* d_ws, size_t ws_size,
                              hipStream_t stream) {
    (void)in_sizes; (void)n_in; (void)out_size; (void)d_ws; (void)ws_size;
    const float* s0 = (const float*)d_in[0];
    const float* s1 = (const float*)d_in[1];
    const float* s2 = (const float*)d_in[2];
    const float* pw = (const float*)d_in[3];
    const float* pb = (const float*)d_in[4];
    const float* lg = (const float*)d_in[5];
    const float* lb = (const float*)d_in[6];
    float* out = (float*)d_out;
    float* wts = out + (size_t)3 * BT * DDIM;   // weights region of the tuple output

    trimodal_weights_kernel<<<BT / 8, 256, 0, stream>>>(s0, s1, s2, wts);
    dim3 g2(BT / 128, DDIM / 128, 3);
    trimodal_gemm_kernel<<<g2, 256, 0, stream>>>(s0, s1, s2, pw, pb, wts, out);
    trimodal_ln_kernel<<<(3 * BT) / 8, 256, 0, stream>>>(out, lg, lb);
}
